// SwinBlock_49976239456751
// MI455X (gfx1250) — compile-verified, matrix operand order checked by reading, not run on hardware
//
#include <hip/hip_runtime.h>
#include <hip/hip_bf16.h>
#include <math.h>

// ---------------- types ----------------
typedef __attribute__((ext_vector_type(16))) _Float16 v16h;
typedef __attribute__((ext_vector_type(8)))  _Float16 v8h;
typedef __attribute__((ext_vector_type(8)))  float    v8f;

// ---------------- problem constants ----------------
static constexpr int BQ   = 4;      // batch
static constexpr int CDIM = 256;    // channels
static constexpr int HH   = 56, WW = 56;
static constexpr int HEADS = 8, HD = 32;
static constexpr int WS = 7, SHIFT = 3;
static constexpr int NWIN = 8;              // nH = nW = 8
static constexpr int LWIN = BQ * NWIN * NWIN;   // 256
static constexpr int NPOS = WS * WS;            // 49
static constexpr int MROWS = LWIN * NPOS;       // 12544 tokens
static constexpr int MLPH = 4 * CDIM;           // 1024
static constexpr int HW = HH * WW;              // 3136

// ---------------- fragment helpers (wave32 WMMA layouts) ----------------
static __device__ __forceinline__ v16h make_frag(const _Float16* p0, const _Float16* p1) {
  v8h lo = *reinterpret_cast<const v8h*>(p0);
  v8h hi = *reinterpret_cast<const v8h*>(p1);
  v16h r;
#pragma unroll
  for (int i = 0; i < 8; ++i) { r[i] = lo[i]; r[i + 8] = hi[i]; }
  return r;
}

static __device__ __forceinline__ v8f wmma_f16(v16h a, v16h b, v8f c) {
  return __builtin_amdgcn_wmma_f32_16x16x32_f16(false, a, false, b, (short)0, c, false, false);
}

// ---------------- f32 -> f16 weight conversion ----------------
__global__ void cvt_f16_kernel(const float* __restrict__ src, _Float16* __restrict__ dst, int n) {
  int i = blockIdx.x * 256 + threadIdx.x;
  if (i < n) dst[i] = (_Float16)src[i];
}

// ---------------- LN1 + roll(-3,-3) + window partition ----------------
__global__ __launch_bounds__(256) void ln1_window_kernel(
    const float* __restrict__ x, const float* __restrict__ w, const float* __restrict__ b,
    _Float16* __restrict__ wnd) {
  int r = blockIdx.x;              // 0..12543
  int c = threadIdx.x;             // 0..255
  int l = r / NPOS, wp = r % NPOS;
  int bb = l >> 6, wy = (l >> 3) & 7, wx = l & 7;
  int iy = wp / WS, ix = wp % WS;
  int y  = (wy * WS + iy + SHIFT) % HH;   // roll(-3): rolled[i] = x[(i+3)%56]
  int xx = (wx * WS + ix + SHIFT) % WW;
  float v = x[((size_t)(bb * CDIM + c)) * HW + y * WW + xx];

  __shared__ float red[256];
  red[c] = v; __syncthreads();
#pragma unroll
  for (int s = 128; s > 0; s >>= 1) { if (c < s) red[c] += red[c + s]; __syncthreads(); }
  float mean = red[0] * (1.0f / 256.0f);
  __syncthreads();
  float d = v - mean;
  red[c] = d * d; __syncthreads();
#pragma unroll
  for (int s = 128; s > 0; s >>= 1) { if (c < s) red[c] += red[c + s]; __syncthreads(); }
  float var = red[0] * (1.0f / 256.0f);
  float ln = d * rsqrtf(var + 1e-5f) * w[c] + b[c];
  wnd[(size_t)r * CDIM + c] = (_Float16)ln;
}

// ---------------- LN2 (in-place f32 + f16 copy) ----------------
__global__ __launch_bounds__(256) void ln2_kernel(
    float* __restrict__ xseq, const float* __restrict__ w, const float* __restrict__ b,
    _Float16* __restrict__ xln) {
  int r = blockIdx.x, c = threadIdx.x;
  float v = xseq[(size_t)r * CDIM + c];
  __shared__ float red[256];
  red[c] = v; __syncthreads();
#pragma unroll
  for (int s = 128; s > 0; s >>= 1) { if (c < s) red[c] += red[c + s]; __syncthreads(); }
  float mean = red[0] * (1.0f / 256.0f);
  __syncthreads();
  float d = v - mean;
  red[c] = d * d; __syncthreads();
#pragma unroll
  for (int s = 128; s > 0; s >>= 1) { if (c < s) red[c] += red[c + s]; __syncthreads(); }
  float var = red[0] * (1.0f / 256.0f);
  float ln = d * rsqrtf(var + 1e-5f) * w[c] + b[c];
  xseq[(size_t)r * CDIM + c] = ln;          // residual base for MLP
  xln[(size_t)r * CDIM + c] = (_Float16)ln; // GEMM3 input
}

// ---------------- register-blocked WMMA GEMM: C = A(M,K) @ W(N,K)^T + epilogue ----------------
// Each wave computes an (MT*16) x (NT*16) tile: MT*NT WMMAs per K-chunk of 32.
// MODE 0: +bias -> f16 out (qkv)
// MODE 1: +bias + window-reverse/roll scatter + shortcut(x) -> f32 xseq
// MODE 2: +bias -> exact GELU -> f16 out (mlp hidden)
// MODE 3: +bias + xseq residual -> d_out in (B,C,H,W) layout
template <int MODE, int MT, int NT>
__global__ __launch_bounds__(32) void gemm_wmma_kernel(
    const _Float16* __restrict__ A, const _Float16* __restrict__ W,
    const float* __restrict__ bias, const float* __restrict__ extra,
    float* __restrict__ fout, _Float16* __restrict__ hout,
    int N, int K) {
  int t = threadIdx.x;
  int lane = t & 15, grp = t >> 4, kb = grp * 8;
  int m0 = blockIdx.x * (MT * 16);
  int n0 = blockIdx.y * (NT * 16);

  const _Float16* arow[MT];
  const _Float16* wrow[NT];
#pragma unroll
  for (int mi = 0; mi < MT; ++mi) arow[mi] = A + (size_t)(m0 + mi * 16 + lane) * K;
#pragma unroll
  for (int ni = 0; ni < NT; ++ni) wrow[ni] = W + (size_t)(n0 + ni * 16 + lane) * K;

  v8f acc[MT][NT];
  v8f zero = {};
#pragma unroll
  for (int mi = 0; mi < MT; ++mi)
#pragma unroll
    for (int ni = 0; ni < NT; ++ni) acc[mi][ni] = zero;

  for (int kc = 0; kc < K; kc += 32) {
    v16h af[MT], bf[NT];
#pragma unroll
    for (int mi = 0; mi < MT; ++mi) af[mi] = make_frag(arow[mi] + kc + kb, arow[mi] + kc + 16 + kb);
#pragma unroll
    for (int ni = 0; ni < NT; ++ni) bf[ni] = make_frag(wrow[ni] + kc + kb, wrow[ni] + kc + 16 + kb);
#pragma unroll
    for (int mi = 0; mi < MT; ++mi)
#pragma unroll
      for (int ni = 0; ni < NT; ++ni) acc[mi][ni] = wmma_f16(af[mi], bf[ni], acc[mi][ni]);
  }

#pragma unroll
  for (int ni = 0; ni < NT; ++ni) {
    int col = n0 + ni * 16 + lane;
    float bcol = bias[col];
#pragma unroll
    for (int mi = 0; mi < MT; ++mi) {
#pragma unroll
      for (int e = 0; e < 8; ++e) {
        int row = m0 + mi * 16 + e + 8 * grp;
        float v = acc[mi][ni][e] + bcol;
        if constexpr (MODE == 0) {
          hout[(size_t)row * N + col] = (_Float16)v;
        } else if constexpr (MODE == 1) {
          int l = row / NPOS, wp = row % NPOS;
          int bb = l >> 6, wy = (l >> 3) & 7, wx = l & 7;
          int iy = wp / WS, ix = wp % WS;
          int y  = (wy * WS + iy + SHIFT) % HH;   // roll back: rolled ys -> y=(ys+3)%56
          int xx = (wx * WS + ix + SHIFT) % WW;
          int p  = y * WW + xx;
          v += extra[((size_t)(bb * CDIM + col)) * HW + p];      // shortcut = x (B,C,H,W)
          fout[((size_t)(bb * HW + p)) * CDIM + col] = v;        // xseq (B,HW,C)
        } else if constexpr (MODE == 2) {
          float g = 0.5f * v * (1.0f + erff(v * 0.70710678118654752f));
          hout[(size_t)row * N + col] = (_Float16)g;
        } else {
          v += extra[(size_t)row * CDIM + col];                  // + LN2(xseq)
          int bb = row / HW, p = row % HW;
          fout[((size_t)(bb * CDIM + col)) * HW + p] = v;        // (B,C,H,W)
        }
      }
    }
  }
}

// ---------------- window attention: one wave per (ltile, head, wpos) ----------------
// V is DMA'd to LDS with gfx1250 async loads at kernel entry, overlapping the
// S = Q@K^T WMMAs and the softmax; s_wait_asynccnt fences before the P@V loop.
__global__ __launch_bounds__(32) void attn_kernel(
    const _Float16* __restrict__ qkv, _Float16* __restrict__ o) {
  int t = threadIdx.x;
  int ltile = blockIdx.x;   // 0..15
  int h     = blockIdx.y;   // 0..7
  int wp    = blockIdx.z;   // 0..48
  int lane = t & 15, grp = t >> 4, kb = grp * 8;

  __shared__ __align__(16) _Float16 Vlds[256 * 32];  // 16 KB: V[m][d]
  __shared__ __align__(16) _Float16 P[16 * 256];     //  8 KB

  // ---- issue async V staging first (overlaps with S computation) ----
  {
    uint32_t vbase = (uint32_t)(uintptr_t)(&Vlds[0]);
    const _Float16* vsrc = qkv + 2 * CDIM + h * HD;
    for (int it = 0; it < 32; ++it) {
      int chunk = it * 32 + t;             // 1024 16-byte chunks total
      int m = chunk >> 2, c = (chunk & 3) * 8;
      const _Float16* g = vsrc + ((size_t)(m * NPOS + wp)) * 768 + c;
      uint32_t laddr = vbase + (uint32_t)(m * 32 + c) * 2;
      asm volatile("global_load_async_to_lds_b128 %0, %1, off"
                   :: "v"(laddr), "v"(g) : "memory");
    }
  }

  // Q A-fragment: rows l = ltile*16 + lane, K-dim = head dim (32)
  size_t qbase = ((size_t)(ltile * 16 + lane) * NPOS + wp) * 768 + h * HD;
  v16h aq = make_frag(qkv + qbase + kb, qkv + qbase + 16 + kb);

  // S = Q @ K^T : 16 tiles, one x32 WMMA each (K-rows load exactly like A-fragments)
  v8f S[16];
#pragma unroll
  for (int mt = 0; mt < 16; ++mt) {
    size_t kaddr = ((size_t)(mt * 16 + lane) * NPOS + wp) * 768 + CDIM + h * HD;
    v16h bk = make_frag(qkv + kaddr + kb, qkv + kaddr + 16 + kb);
    v8f z = {};
    S[mt] = wmma_f16(aq, bk, z);
  }

  const float scale = 0.17677669529663687f;  // 1/sqrt(32)

  // softmax per row r = e + 8*grp over all 256 columns (16 lanes-in-group x 16 tiles)
  float rmax[8], rsum[8];
#pragma unroll
  for (int e = 0; e < 8; ++e) {
    float m = -3.4e38f;
#pragma unroll
    for (int mt = 0; mt < 16; ++mt) { S[mt][e] *= scale; m = fmaxf(m, S[mt][e]); }
    rmax[e] = m;
  }
#pragma unroll
  for (int off = 1; off < 16; off <<= 1)
#pragma unroll
    for (int e = 0; e < 8; ++e) rmax[e] = fmaxf(rmax[e], __shfl_xor(rmax[e], off, 32));
#pragma unroll
  for (int e = 0; e < 8; ++e) {
    float s = 0.0f;
#pragma unroll
    for (int mt = 0; mt < 16; ++mt) { float ev = __expf(S[mt][e] - rmax[e]); S[mt][e] = ev; s += ev; }
    rsum[e] = s;
  }
#pragma unroll
  for (int off = 1; off < 16; off <<= 1)
#pragma unroll
    for (int e = 0; e < 8; ++e) rsum[e] += __shfl_xor(rsum[e], off, 32);

  // stage P through LDS to convert C-layout -> A-layout
#pragma unroll
  for (int e = 0; e < 8; ++e) {
    float inv = 1.0f / rsum[e];
    int row = e + 8 * grp;
#pragma unroll
    for (int mt = 0; mt < 16; ++mt)
      P[row * 256 + mt * 16 + lane] = (_Float16)(S[mt][e] * inv);
  }
  __syncthreads();

  // make sure async V staging has landed in LDS
  asm volatile("s_wait_asynccnt 0" ::: "memory");

  // O = P(16x256) @ V(256x32): K-chunks of 32, two N-tiles (d=0..15, 16..31)
  v8f o0 = {}, o1 = {};
  for (int kc = 0; kc < 8; ++kc) {
    const _Float16* prow = &P[lane * 256 + kc * 32];
    v16h ap = make_frag(prow + kb, prow + 16 + kb);
    v16h bv0, bv1;
#pragma unroll
    for (int i = 0; i < 16; ++i) {
      int kk = (i < 8) ? (kb + i) : (16 + kb + (i - 8));
      int vrow = (kc * 32 + kk) * 32;
      bv0[i] = Vlds[vrow + lane];
      bv1[i] = Vlds[vrow + 16 + lane];
    }
    o0 = wmma_f16(ap, bv0, o0);
    o1 = wmma_f16(ap, bv1, o1);
  }

  // store O to (L, N, C) f16
#pragma unroll
  for (int e = 0; e < 8; ++e) {
    int l = ltile * 16 + e + 8 * grp;
    size_t base = ((size_t)l * NPOS + wp) * CDIM + h * HD;
    o[base + lane]      = (_Float16)o0[e];
    o[base + 16 + lane] = (_Float16)o1[e];
  }
}

// ---------------- host launcher ----------------
extern "C" void kernel_launch(void* const* d_in, const int* in_sizes, int n_in,
                              void* d_out, int out_size, void* d_ws, size_t ws_size,
                              hipStream_t stream) {
  const float* x          = (const float*)d_in[0];
  const float* ln1_w      = (const float*)d_in[1];
  const float* ln1_b      = (const float*)d_in[2];
  const float* in_proj_w  = (const float*)d_in[3];   // (768,256)
  const float* in_proj_b  = (const float*)d_in[4];
  const float* out_proj_w = (const float*)d_in[5];   // (256,256)
  const float* out_proj_b = (const float*)d_in[6];
  const float* ln2_w      = (const float*)d_in[7];
  const float* ln2_b      = (const float*)d_in[8];
  const float* mlp_w1     = (const float*)d_in[9];   // (1024,256)
  const float* mlp_b1     = (const float*)d_in[10];
  const float* mlp_w2     = (const float*)d_in[11];  // (256,1024)
  const float* mlp_b2     = (const float*)d_in[12];
  float* out = (float*)d_out;

  char* ws = (char*)d_ws;
  // arena (bytes, all offsets 256-aligned); total = 52,953,088
  _Float16* w_in  = (_Float16*)(ws + 0);              // 768*256  -> 393,216 B
  _Float16* w_out = (_Float16*)(ws + 393216);         // 256*256  -> 131,072 B
  _Float16* w_m1  = (_Float16*)(ws + 524288);         // 1024*256 -> 524,288 B
  _Float16* w_m2  = (_Float16*)(ws + 1048576);        // 256*1024 -> 524,288 B
  _Float16* wnd   = (_Float16*)(ws + 1572864);        // 12544*256 f16 -> 6,422,528 B
  _Float16* qkvh  = (_Float16*)(ws + 7995392);        // 12544*768 f16 -> 19,267,584 B
  _Float16* hidden= (_Float16*)(ws + 1572864);        // reuse wnd+qkv region (25,690,112 B)
  _Float16* of16  = (_Float16*)(ws + 27262976);       // 12544*256 f16 -> 6,422,528 B
  float*    xseq  = (float*)   (ws + 33685504);       // 12544*256 f32 -> 12,845,056 B
  _Float16* xln   = (_Float16*)(ws + 46530560);       // 12544*256 f16 -> 6,422,528 B

  // 1) weights -> f16 (row-major (N,K) kept as-is)
  cvt_f16_kernel<<<(3*CDIM*CDIM + 255)/256, 256, 0, stream>>>(in_proj_w,  w_in,  3*CDIM*CDIM);
  cvt_f16_kernel<<<(CDIM*CDIM   + 255)/256, 256, 0, stream>>>(out_proj_w, w_out, CDIM*CDIM);
  cvt_f16_kernel<<<(MLPH*CDIM   + 255)/256, 256, 0, stream>>>(mlp_w1,     w_m1,  MLPH*CDIM);
  cvt_f16_kernel<<<(CDIM*MLPH   + 255)/256, 256, 0, stream>>>(mlp_w2,     w_m2,  CDIM*MLPH);

  // 2) LN1 + roll + window partition
  ln1_window_kernel<<<MROWS, 256, 0, stream>>>(x, ln1_w, ln1_b, wnd);

  // 3) QKV projection: (12544,256) @ (256,768), 64x64 tiles/wave
  gemm_wmma_kernel<0,4,4><<<dim3(MROWS/64, (3*CDIM)/64), 32, 0, stream>>>(
      wnd, w_in, in_proj_b, nullptr, nullptr, qkvh, 3*CDIM, CDIM);

  // 4) window attention (392 independent 256x256x32 attentions)
  attn_kernel<<<dim3(LWIN/16, HEADS, NPOS), 32, 0, stream>>>(qkvh, of16);

  // 5) out projection + window reverse + roll + residual -> xseq (B,HW,C), 32x32 tiles
  gemm_wmma_kernel<1,2,2><<<dim3(MROWS/32, CDIM/32), 32, 0, stream>>>(
      of16, w_out, out_proj_b, x, xseq, nullptr, CDIM, CDIM);

  // 6) LN2 (overwrites xseq with LN output; f16 copy for GEMM3)
  ln2_kernel<<<MROWS, 256, 0, stream>>>(xseq, ln2_w, ln2_b, xln);

  // 7) MLP up + GELU: (12544,256) @ (256,1024), 64x64 tiles/wave
  gemm_wmma_kernel<2,4,4><<<dim3(MROWS/64, MLPH/64), 32, 0, stream>>>(
      xln, w_m1, mlp_b1, nullptr, nullptr, hidden, MLPH, CDIM);

  // 8) MLP down + residual -> d_out in (B,C,H,W), 32x32 tiles
  gemm_wmma_kernel<3,2,2><<<dim3(MROWS/32, CDIM/32), 32, 0, stream>>>(
      hidden, w_m2, mlp_b2, xseq, out, nullptr, CDIM, MLPH);
}
